// linear_attention_19250043421067
// MI455X (gfx1250) — compile-verified, matrix operand order checked by reading, not run on hardware
//
#include <hip/hip_runtime.h>
#include <hip/hip_bf16.h>

typedef __attribute__((ext_vector_type(16))) __bf16 v16bf;
typedef __attribute__((ext_vector_type(8)))  __bf16 v8bf;
typedef __attribute__((ext_vector_type(8)))  float  v8f;

#define D_DIM 1024
#define S_LEN 2048
#define B_SZ  4
#define M_TOT (B_SZ * S_LEN)          /* 8192 token rows */
#define SCALE_LOGIT 0.015625f         /* TAU / sqrt(D) = 0.5 / 32 */

// ---------------- fragment loaders (CDNA5 16-bit A/B layout) ----------------
// Per ISA 05_wmma: lane L holds row/col (L&15); k-halves at kbase=(L>>4)*8:
// halfs[0..7] = k kbase..kbase+7, halfs[8..15] = k kbase+16..kbase+23.
// Both halves are k-contiguous in memory -> two 16B loads.
static __device__ inline v16bf frag_from_bf16(const __bf16* p) {
  v8bf a = *(const v8bf*)(p);
  v8bf b = *(const v8bf*)(p + 16);
  return __builtin_shufflevector(a, b, 0,1,2,3,4,5,6,7,8,9,10,11,12,13,14,15);
}

static __device__ inline v8f wmma_bf16(v16bf a, v16bf b, v8f c) {
  // (neg_a, A, neg_b, B, c_mod, C, reuse_a, reuse_b)
  return __builtin_amdgcn_wmma_f32_16x16x32_bf16(false, a, false, b, (short)0, c,
                                                 false, false);
}

// ---------------- f32 -> bf16 bulk convert (memory-bound, one pass) --------
__global__ __launch_bounds__(256) void cvt_f32_bf16_kernel(
    const float* __restrict__ src, __bf16* __restrict__ dst, int n8)
{
  const int i = blockIdx.x * 256 + threadIdx.x;
  if (i < n8) {
    v8f a = *(const v8f*)(src + (size_t)i * 8);
    v8bf o;
#pragma unroll
    for (int j = 0; j < 8; ++j) o[j] = (__bf16)a[j];
    *(v8bf*)(dst + (size_t)i * 8) = o;
  }
}

// ---------------- bf16 GEMM: C = A @ W^T + bias ----------------------------
// A: [8192,1024] bf16, W: [1024,1024] bf16 (both k-contiguous), bias f32.
// Each wave: 32 rows x 64 cols (2 m-tiles x 4 n-tiles) -> B frags reused 2x.
// c_mode: 0 -> bf16 row-major; 1 -> bf16 transposed [B][D][S]; 2 -> f32 row-major
__global__ __launch_bounds__(256) void gemm_bf16_kernel(
    const __bf16* __restrict__ A, const __bf16* __restrict__ W,
    const float* __restrict__ bias, void* __restrict__ Cout, int c_mode)
{
  const int lane = threadIdx.x & 31;
  const int wid  = blockIdx.x * 8 + (threadIdx.x >> 5);
  const int m0 = (wid >> 4) << 5;   // 256 m-supertiles of 32 rows
  const int n0 = (wid & 15) << 6;   // 16 n-groups of 64 cols
  const int lm = lane & 15;
  const int lh = lane >> 4;

  v8f acc[2][4];
#pragma unroll
  for (int nt = 0; nt < 4; ++nt) {
    const float bv = bias[n0 + nt*16 + lm];
#pragma unroll
    for (int r = 0; r < 8; ++r) { acc[0][nt][r] = bv; acc[1][nt][r] = bv; }
  }

  const __bf16* A0 = A + (size_t)(m0 + lm) * D_DIM;
  const __bf16* A1 = A0 + (size_t)16 * D_DIM;

  for (int k0 = 0; k0 < D_DIM; k0 += 32) {
    const int ka = k0 + lh * 8;
    v16bf a0 = frag_from_bf16(A0 + ka);
    v16bf a1 = frag_from_bf16(A1 + ka);
#pragma unroll
    for (int nt = 0; nt < 4; ++nt) {
      v16bf bfrag = frag_from_bf16(W + (size_t)(n0 + nt*16 + lm) * D_DIM + ka);
      acc[0][nt] = wmma_bf16(a0, bfrag, acc[0][nt]);
      acc[1][nt] = wmma_bf16(a1, bfrag, acc[1][nt]);
    }
  }

#pragma unroll
  for (int mt = 0; mt < 2; ++mt) {
#pragma unroll
    for (int nt = 0; nt < 4; ++nt) {
#pragma unroll
      for (int r = 0; r < 8; ++r) {
        const int m = m0 + mt*16 + r + lh*8;   // C layout: M = r + 8*(lane/16)
        const int j = n0 + nt*16 + lm;         //           N = lane%16
        const float v = acc[mt][nt][r];
        if (c_mode == 0) {
          ((__bf16*)Cout)[(size_t)m * D_DIM + j] = (__bf16)v;
        } else if (c_mode == 1) {
          const int bb = m >> 11, s = m & (S_LEN - 1);
          ((__bf16*)Cout)[((size_t)bb * D_DIM + j) * S_LEN + s] = (__bf16)v;
        } else {
          ((float*)Cout)[(size_t)m * D_DIM + j] = v;
        }
      }
    }
  }
}

// ------------- async staging of one 32-key K block slice into LDS ----------
// Each wave copies its own k-slice [wv*128, wv*128+128) for all 32 keys:
// 512 x 16B chunks, 16 per lane, via global_load_async_to_lds_b128 (ASYNCcnt).
static __device__ inline void async_stage_kblock(const __bf16* Kb, int kb,
                                                 int wv, int lane, __bf16* dst)
{
#pragma unroll
  for (int i = 0; i < 16; ++i) {
    const int c   = i * 32 + lane;       // chunk id within wave's 8KB region
    const int key = c >> 4;              // 16 chunks (256B) per key row
    const int el  = wv * 128 + (c & 15) * 8;
    const unsigned long long g =
        (unsigned long long)(uintptr_t)(Kb + (size_t)(kb + key) * D_DIM + el);
    const unsigned int l =
        (unsigned int)(uintptr_t)(dst + (size_t)key * D_DIM + el);
    asm volatile("global_load_async_to_lds_b128 %0, %1, off"
                 :: "v"(l), "v"(g) : "memory");
  }
}

// ---------------- flash attention (causal, online softmax) -----------------
// One workgroup (8 waves) per (batch, 16-row query tile).
// Wave w owns D-slice [w*128, w*128+128) of the PV accumulator and k-slice
// [w*128, w*128+128) of the QK^T reduction (partials reduced via ds_add_f32).
// K blocks are double-buffered in LDS via async-to-LDS copies.
__global__ __launch_bounds__(256) void flash_attn_kernel(
    const __bf16* __restrict__ Q, const __bf16* __restrict__ K,
    const __bf16* __restrict__ Vt, __bf16* __restrict__ Z)
{
  __shared__ __attribute__((aligned(16))) __bf16 s_k[2][32 * D_DIM];  // 128 KB
  __shared__ __attribute__((aligned(16))) float  s_score[16 * 32];
  __shared__ __attribute__((aligned(64))) __bf16 s_p[16 * 32];
  __shared__ float s_m[16], s_l[16], s_alpha[16];

  const int tid  = threadIdx.x;
  const int lane = tid & 31;
  const int wv   = tid >> 5;               // wave 0..7
  const int lm   = lane & 15;
  const int lh   = lane >> 4;
  const int b    = blockIdx.x >> 7;        // 4 batches
  const int q0   = (blockIdx.x & 127) << 4;

  const __bf16* Qb = Q  + ((size_t)(b * S_LEN + q0)) * D_DIM;
  const __bf16* Kb = K  + ((size_t)(b * S_LEN)) * D_DIM;
  const __bf16* Vb = Vt + ((size_t)b) * D_DIM * S_LEN;

  // preload this wave's Q fragments (16 rows x 128 k)
  v16bf qf[4];
#pragma unroll
  for (int f = 0; f < 4; ++f)
    qf[f] = frag_from_bf16(Qb + (size_t)lm * D_DIM + wv*128 + f*32 + lh*8);

  v8f acc[8];
#pragma unroll
  for (int nt = 0; nt < 8; ++nt)
#pragma unroll
    for (int r = 0; r < 8; ++r) acc[nt][r] = 0.f;

  if (tid < 16) { s_m[tid] = -3.0e38f; s_l[tid] = 0.f; }

  // kick off async staging of the first K block
  async_stage_kblock(Kb, 0, wv, lane, &s_k[0][0]);
  int buf = 0;
  __syncthreads();

  const int kend = q0 + 15;
  for (int kb = 0; kb <= kend; kb += 32) {
    s_score[tid]       = 0.f;
    s_score[tid + 256] = 0.f;

    // stage the next block into the other buffer while we compute this one
    const bool has_next = (kb + 32) <= kend;
    if (has_next) {
      async_stage_kblock(Kb, kb + 32, wv, lane, &s_k[buf ^ 1][0]);
      asm volatile("s_wait_asynccnt 0x10" ::: "memory");  // block kb landed
    } else {
      asm volatile("s_wait_asynccnt 0x0"  ::: "memory");
    }
    __syncthreads();

    // partial Q*K^T over this wave's 128-wide k slice, two 16-key subtiles
#pragma unroll
    for (int st = 0; st < 2; ++st) {
      const __bf16* Kr = &s_k[buf][(size_t)(st*16 + lm) * D_DIM + wv*128 + lh*8];
      v8f c = {};
#pragma unroll
      for (int f = 0; f < 4; ++f)
        c = wmma_bf16(qf[f], frag_from_bf16(Kr + f*32), c);
#pragma unroll
      for (int r = 0; r < 8; ++r)
        atomicAdd(&s_score[(r + lh*8) * 32 + st*16 + lm], c[r]);
    }
    // speculative prefetch of the next V block rows (dropped if OOB)
    __builtin_prefetch(Vb + (size_t)(wv*128 + lm) * S_LEN + kb + 32, 0, 1);
    __syncthreads();

    // online softmax: one lane per query row
    if (tid < 16) {
      const int q = q0 + tid;
      const float mold = s_m[tid];
      float mnew = mold;
      float sj[32];
#pragma unroll
      for (int j = 0; j < 32; ++j) {
        float s = s_score[tid*32 + j] * SCALE_LOGIT;
        if (kb + j > q) s = -5.0e8f;   // masked_fill(-1e9) * TAU
        sj[j] = s;
        mnew = fmaxf(mnew, s);
      }
      const float alpha = __expf(mold - mnew);
      float lsum = 0.f;
#pragma unroll
      for (int j = 0; j < 32; ++j) {
        const float p = __expf(sj[j] - mnew);
        lsum += p;
        s_p[tid*32 + j] = (__bf16)p;
      }
      s_m[tid]     = mnew;
      s_l[tid]     = s_l[tid] * alpha + lsum;
      s_alpha[tid] = alpha;
    }
    __syncthreads();

    // rescale accumulators by alpha(row), then acc += P * V(slice)
    float alr[8];
#pragma unroll
    for (int r = 0; r < 8; ++r) alr[r] = s_alpha[r + lh*8];
#pragma unroll
    for (int nt = 0; nt < 8; ++nt)
#pragma unroll
      for (int r = 0; r < 8; ++r) acc[nt][r] *= alr[r];

    v16bf pf = frag_from_bf16(&s_p[lm*32 + lh*8]);   // P is 16x32: one A frag
#pragma unroll
    for (int nt = 0; nt < 8; ++nt) {
      const int d = wv*128 + nt*16 + lm;             // B frag: Vt[d] k-contig
      v16bf vfrag = frag_from_bf16(Vb + (size_t)d * S_LEN + kb + lh*8);
      acc[nt] = wmma_bf16(pf, vfrag, acc[nt]);
    }
    buf ^= 1;
  }

  // normalize by row sums and emit z in bf16 row-major
  float linv[8];
#pragma unroll
  for (int r = 0; r < 8; ++r) linv[r] = 1.0f / s_l[r + lh*8];
#pragma unroll
  for (int nt = 0; nt < 8; ++nt) {
#pragma unroll
    for (int r = 0; r < 8; ++r) {
      const int m = q0 + r + lh*8;
      const int d = wv*128 + nt*16 + lm;
      Z[((size_t)(b * S_LEN + m)) * D_DIM + d] = (__bf16)(acc[nt][r] * linv[r]);
    }
  }
}

// ---------------------------------------------------------------------------
extern "C" void kernel_launch(void* const* d_in, const int* in_sizes, int n_in,
                              void* d_out, int out_size, void* d_ws, size_t ws_size,
                              hipStream_t stream) {
  (void)in_sizes; (void)n_in; (void)out_size; (void)ws_size;
  const float* node  = (const float*)d_in[0];
  const float* label = (const float*)d_in[1];
  const float* Wq = (const float*)d_in[2];
  const float* bq = (const float*)d_in[3];
  const float* Wk = (const float*)d_in[4];
  const float* bk = (const float*)d_in[5];
  const float* Wv = (const float*)d_in[6];
  const float* bv = (const float*)d_in[7];
  const float* Wp = (const float*)d_in[8];
  const float* bp = (const float*)d_in[9];

  char* ws = (char*)d_ws;
  const size_t seg  = (size_t)M_TOT * D_DIM * sizeof(__bf16);   // 16 MiB
  const size_t wseg = (size_t)D_DIM * D_DIM * sizeof(__bf16);   //  2 MiB
  __bf16* Qb  = (__bf16*)(ws);
  __bf16* Kb  = (__bf16*)(ws + seg);
  __bf16* Vtb = (__bf16*)(ws + 2 * seg);
  __bf16* Zb  = (__bf16*)(ws + 3 * seg);
  __bf16* Nb  = (__bf16*)(ws + 4 * seg);          // node_emb bf16
  __bf16* Lb  = (__bf16*)(ws + 5 * seg);          // label_emb bf16
  __bf16* Wqb = (__bf16*)(ws + 6 * seg);
  __bf16* Wkb = (__bf16*)(ws + 6 * seg + wseg);
  __bf16* Wvb = (__bf16*)(ws + 6 * seg + 2 * wseg);
  __bf16* Wpb = (__bf16*)(ws + 6 * seg + 3 * wseg);

  // one-shot f32->bf16 conversions (memory bound; ~5us of HBM traffic total)
  const int emb8 = M_TOT * D_DIM / 8;     // 1,048,576 v8 chunks
  const int w8   = D_DIM * D_DIM / 8;     //   131,072 v8 chunks
  cvt_f32_bf16_kernel<<<dim3(emb8 / 256), dim3(256), 0, stream>>>(node,  Nb,  emb8);
  cvt_f32_bf16_kernel<<<dim3(emb8 / 256), dim3(256), 0, stream>>>(label, Lb,  emb8);
  cvt_f32_bf16_kernel<<<dim3(w8 / 256),   dim3(256), 0, stream>>>(Wq,    Wqb, w8);
  cvt_f32_bf16_kernel<<<dim3(w8 / 256),   dim3(256), 0, stream>>>(Wk,    Wkb, w8);
  cvt_f32_bf16_kernel<<<dim3(w8 / 256),   dim3(256), 0, stream>>>(Wv,    Wvb, w8);
  cvt_f32_bf16_kernel<<<dim3(w8 / 256),   dim3(256), 0, stream>>>(Wp,    Wpb, w8);

  const dim3 gemm_grid(512), gemm_blk(256);   // 4096 waves: 256 mtiles x 16 ngroups
  gemm_bf16_kernel<<<gemm_grid, gemm_blk, 0, stream>>>(Lb, Wqb, bq, Qb, 0);
  gemm_bf16_kernel<<<gemm_grid, gemm_blk, 0, stream>>>(Nb, Wkb, bk, Kb, 0);
  gemm_bf16_kernel<<<gemm_grid, gemm_blk, 0, stream>>>(Nb, Wvb, bv, Vtb, 1);
  flash_attn_kernel<<<dim3(B_SZ * (S_LEN / 16)), dim3(256), 0, stream>>>(Qb, Kb, Vtb, Zb);
  gemm_bf16_kernel<<<gemm_grid, gemm_blk, 0, stream>>>(Zb, Wpb, bp, d_out, 2);
}
